// Block_45853070852659
// MI455X (gfx1250) — compile-verified, matrix operand order checked by reading, not run on hardware
//
#include <hip/hip_runtime.h>
#include <hip/hip_bf16.h>
#include <math.h>
#include <stdint.h>

typedef __bf16 bf16;
typedef __attribute__((ext_vector_type(16))) __bf16 v16bf;
typedef __attribute__((ext_vector_type(8)))  __bf16 v8bf;
typedef __attribute__((ext_vector_type(8)))  float  v8f;
typedef __attribute__((ext_vector_type(4)))  int    v4i;

#define WMMA_BF16(a, b, c) \
  __builtin_amdgcn_wmma_f32_16x16x32_bf16(false, (a), false, (b), (short)0, (c), false, false)

#ifndef __has_builtin
#define __has_builtin(x) 0
#endif
#if __has_builtin(__builtin_amdgcn_global_load_async_to_lds_b128) && \
    __has_builtin(__builtin_amdgcn_s_wait_asynccnt)
#define USE_ASYNC_LDS 1
#else
#define USE_ASYNC_LDS 0
#endif

#if USE_ASYNC_LDS
// 16B global -> LDS async copy (per-lane), tracked by ASYNCcnt.
// Builtin signature (from clang diagnostic): (v4i as1*, v4i as3*, Imm offset, Imm cpol)
#define ASYNC_CP16(gptr, lptr)                                                   \
  __builtin_amdgcn_global_load_async_to_lds_b128(                                \
      (__attribute__((address_space(1))) v4i*)(uintptr_t)(gptr),                 \
      (__attribute__((address_space(3))) v4i*)(uint32_t)(uintptr_t)(lptr), 0, 0)
#endif

// Problem constants (match reference)
constexpr int kB  = 2;
constexpr int kS  = 1024;
constexpr int kH  = 1024;
constexpr int kNH = 8;
constexpr int kHD = 128;
constexpr int kE  = 8;
constexpr int kMI = 1024;
constexpr int kSI = 2048;
constexpr int kT  = kB * kS;   // 2048 tokens

__device__ __forceinline__ v8f zero8() {
  v8f z = {0.f, 0.f, 0.f, 0.f, 0.f, 0.f, 0.f, 0.f};
  return z;
}

// Load a 16x32 bf16 fragment (A-layout per CDNA5 ISA 7.12.2) from a row-major
// matrix with leading dimension ld. Works for both A (MxK) and B^T (NxK),
// and for both global and LDS pointers.
__device__ __forceinline__ v16bf load_frag(const bf16* __restrict__ base, int ld) {
  const int lane = threadIdx.x & 31;
  const int hg   = lane >> 4;       // lane group: 0 -> K 0..7/16..23, 1 -> K 8..15/24..31
  const int r    = lane & 15;       // row within tile
  const bf16* p = base + (size_t)r * ld + hg * 8;
  v8bf lo = *reinterpret_cast<const v8bf*>(p);
  v8bf hi = *reinterpret_cast<const v8bf*>(p + 16);
  return __builtin_shufflevector(lo, hi, 0, 1, 2, 3, 4, 5, 6, 7, 8, 9, 10, 11, 12, 13, 14, 15);
}

// ---------------------------------------------------------------------------
// LDS-staged WMMA GEMM:  C[M,N] = A[M,K] * W[N,K]^T (+ R)   (A,W bf16; C,R f32)
// Block = 256 threads = 8 waves arranged 4(M) x 2(N); wave tile 32x64.
// Block tile 128x128; K-step 32; double-buffered LDS (padded rows: stride 40
// bf16 = 80B -> conflict-free ds_load_b128 across 16 row-parallel lanes).
// cntPtr: optional device row-limit (grouped MoE GEMM) -> block-uniform exit.
// ---------------------------------------------------------------------------
#define LDS_LD 40   // padded leading dim (elements) for 32-wide K tiles

__global__ void __launch_bounds__(256)
gemm_bf16_kernel(const bf16* __restrict__ A, const bf16* __restrict__ W,
                 float* __restrict__ C, const float* __restrict__ R,
                 int M, int N, int K, const int* __restrict__ cntPtr) {
  const int m0b = blockIdx.y * 128;
  const int n0b = blockIdx.x * 128;
  if (m0b >= M || n0b >= N) return;              // block-uniform
  if (cntPtr && m0b >= *cntPtr) return;          // block-uniform (device count)

  __shared__ bf16 ldsA[2][128][LDS_LD];
  __shared__ bf16 ldsB[2][128][LDS_LD];

  const int tid  = threadIdx.x;
  const int wave = tid >> 5;
  const int wm = wave >> 1;            // 0..3
  const int wn = wave & 1;             // 0..1

  // copy indexing: 512 chunks of 16B per 128x32 tile; this thread owns 2.
  const int c0 = tid, c1 = tid + 256;
  const int rowA0 = c0 >> 2, qA0 = c0 & 3;
  const int rowA1 = c1 >> 2, qA1 = c1 & 3;

  v8f acc[2][4];
#pragma unroll
  for (int i = 0; i < 2; ++i)
#pragma unroll
    for (int j = 0; j < 4; ++j) acc[i][j] = zero8();

  const int nsteps = K >> 5;
  int buf = 0;

#if USE_ASYNC_LDS
  // ---- prologue: async-copy tile 0 into buffer 0 ----
  {
    const int k0 = 0;
    ASYNC_CP16(A + (size_t)(m0b + rowA0) * K + k0 + qA0 * 8, &ldsA[0][rowA0][qA0 * 8]);
    ASYNC_CP16(A + (size_t)(m0b + rowA1) * K + k0 + qA1 * 8, &ldsA[0][rowA1][qA1 * 8]);
    ASYNC_CP16(W + (size_t)(n0b + rowA0) * K + k0 + qA0 * 8, &ldsB[0][rowA0][qA0 * 8]);
    ASYNC_CP16(W + (size_t)(n0b + rowA1) * K + k0 + qA1 * 8, &ldsB[0][rowA1][qA1 * 8]);
    __builtin_amdgcn_s_wait_asynccnt(0);
  }
  __syncthreads();

  for (int step = 0; step < nsteps; ++step) {
    const bool more = (step + 1) < nsteps;
    if (more) {   // overlap next tile's DMA with this tile's WMMA
      const int k0 = (step + 1) << 5;
      const int nb = buf ^ 1;
      ASYNC_CP16(A + (size_t)(m0b + rowA0) * K + k0 + qA0 * 8, &ldsA[nb][rowA0][qA0 * 8]);
      ASYNC_CP16(A + (size_t)(m0b + rowA1) * K + k0 + qA1 * 8, &ldsA[nb][rowA1][qA1 * 8]);
      ASYNC_CP16(W + (size_t)(n0b + rowA0) * K + k0 + qA0 * 8, &ldsB[nb][rowA0][qA0 * 8]);
      ASYNC_CP16(W + (size_t)(n0b + rowA1) * K + k0 + qA1 * 8, &ldsB[nb][rowA1][qA1 * 8]);
    }
    v16bf a0 = load_frag(&ldsA[buf][wm * 32][0], LDS_LD);
    v16bf a1 = load_frag(&ldsA[buf][wm * 32 + 16][0], LDS_LD);
#pragma unroll
    for (int j = 0; j < 4; ++j) {
      v16bf bj = load_frag(&ldsB[buf][wn * 64 + j * 16][0], LDS_LD);
      acc[0][j] = WMMA_BF16(a0, bj, acc[0][j]);
      acc[1][j] = WMMA_BF16(a1, bj, acc[1][j]);
    }
    if (more) {
      __builtin_amdgcn_s_wait_asynccnt(0);
      __syncthreads();
      buf ^= 1;
    }
  }
#else
  // ---- fallback: register-staged double buffering ----
  v8bf ra0, ra1, rb0, rb1;
  {
    const int k0 = 0;
    ra0 = *reinterpret_cast<const v8bf*>(A + (size_t)(m0b + rowA0) * K + k0 + qA0 * 8);
    ra1 = *reinterpret_cast<const v8bf*>(A + (size_t)(m0b + rowA1) * K + k0 + qA1 * 8);
    rb0 = *reinterpret_cast<const v8bf*>(W + (size_t)(n0b + rowA0) * K + k0 + qA0 * 8);
    rb1 = *reinterpret_cast<const v8bf*>(W + (size_t)(n0b + rowA1) * K + k0 + qA1 * 8);
  }
  *reinterpret_cast<v8bf*>(&ldsA[0][rowA0][qA0 * 8]) = ra0;
  *reinterpret_cast<v8bf*>(&ldsA[0][rowA1][qA1 * 8]) = ra1;
  *reinterpret_cast<v8bf*>(&ldsB[0][rowA0][qA0 * 8]) = rb0;
  *reinterpret_cast<v8bf*>(&ldsB[0][rowA1][qA1 * 8]) = rb1;
  __syncthreads();

  for (int step = 0; step < nsteps; ++step) {
    const bool more = (step + 1) < nsteps;
    if (more) {   // issue next tile's global loads before consuming LDS
      const int k0 = (step + 1) << 5;
      ra0 = *reinterpret_cast<const v8bf*>(A + (size_t)(m0b + rowA0) * K + k0 + qA0 * 8);
      ra1 = *reinterpret_cast<const v8bf*>(A + (size_t)(m0b + rowA1) * K + k0 + qA1 * 8);
      rb0 = *reinterpret_cast<const v8bf*>(W + (size_t)(n0b + rowA0) * K + k0 + qA0 * 8);
      rb1 = *reinterpret_cast<const v8bf*>(W + (size_t)(n0b + rowA1) * K + k0 + qA1 * 8);
    }
    v16bf a0 = load_frag(&ldsA[buf][wm * 32][0], LDS_LD);
    v16bf a1 = load_frag(&ldsA[buf][wm * 32 + 16][0], LDS_LD);
#pragma unroll
    for (int j = 0; j < 4; ++j) {
      v16bf bj = load_frag(&ldsB[buf][wn * 64 + j * 16][0], LDS_LD);
      acc[0][j] = WMMA_BF16(a0, bj, acc[0][j]);
      acc[1][j] = WMMA_BF16(a1, bj, acc[1][j]);
    }
    if (more) {
      const int nb = buf ^ 1;
      *reinterpret_cast<v8bf*>(&ldsA[nb][rowA0][qA0 * 8]) = ra0;
      *reinterpret_cast<v8bf*>(&ldsA[nb][rowA1][qA1 * 8]) = ra1;
      *reinterpret_cast<v8bf*>(&ldsB[nb][rowA0][qA0 * 8]) = rb0;
      *reinterpret_cast<v8bf*>(&ldsB[nb][rowA1][qA1 * 8]) = rb1;
      __syncthreads();
      buf = nb;
    }
  }
#endif

  // ---- epilogue ----
  const int lane = tid & 31;
  const int hg = lane >> 4;
  const int col = lane & 15;
  const int m0w = m0b + wm * 32;
  const int n0w = n0b + wn * 64;
#pragma unroll
  for (int i = 0; i < 2; ++i) {
#pragma unroll
    for (int j = 0; j < 4; ++j) {
#pragma unroll
      for (int r = 0; r < 8; ++r) {
        const int m = m0w + i * 16 + r + 8 * hg;
        const int n = n0w + j * 16 + col;
        const size_t idx = (size_t)m * N + n;
        C[idx] = acc[i][j][r] + (R ? R[idx] : 0.f);
      }
    }
  }
}

// ---------------------------------------------------------------------------
// Flash attention (causal, online softmax), one wave per 16-row Q tile.
// Q,K bf16 (token-major, stride H); Vt bf16 (b,h,d,s); O f32 (token-major).
// ---------------------------------------------------------------------------
__global__ void __launch_bounds__(256)
flash_attn_kernel(const bf16* __restrict__ Q, const bf16* __restrict__ K,
                  const bf16* __restrict__ Vt, float* __restrict__ O) {
  const int b = blockIdx.z;
  const int h = blockIdx.y;
  const int wave = threadIdx.x >> 5;
  const int q0 = blockIdx.x * 128 + wave * 16;
  const int lane = threadIdx.x & 31;
  const int hg = lane >> 4;
  const int col = lane & 15;

  const bf16* Qt = Q + ((size_t)(b * kS + q0)) * kH + (size_t)h * kHD;
  const bf16* Kt = K + ((size_t)(b * kS)) * kH + (size_t)h * kHD;
  const bf16* Vh = Vt + ((size_t)(b * kNH + h)) * kHD * kS;   // (d, s), stride kS

  v16bf qa[4];
#pragma unroll
  for (int j = 0; j < 4; ++j) qa[j] = load_frag(Qt + j * 32, kH);

  v8f o[8];
#pragma unroll
  for (int j = 0; j < 8; ++j) o[j] = zero8();
  float mrow[8], lrow[8];
#pragma unroll
  for (int r = 0; r < 8; ++r) { mrow[r] = -INFINITY; lrow[r] = 0.f; }

  __shared__ bf16 pshm[8][16][32];
  const float scale = 0.08838834764831845f;   // 1/sqrt(128)

  const int kend = q0 + 16;                   // keys needed: <= q0+15
  for (int ks = 0; ks < kend; ks += 32) {
    // ---- scores: 16 q-rows x 32 keys ----
    v8f s0 = zero8(), s1 = zero8();
#pragma unroll
    for (int j = 0; j < 4; ++j) {
      v16bf kf0 = load_frag(Kt + (size_t)ks * kH + j * 32, kH);
      v16bf kf1 = load_frag(Kt + (size_t)(ks + 16) * kH + j * 32, kH);
      s0 = WMMA_BF16(qa[j], kf0, s0);
      s1 = WMMA_BF16(qa[j], kf1, s1);
    }

    // ---- online softmax (per-row over the 16-lane group) ----
    float alphar[8];
#pragma unroll
    for (int r = 0; r < 8; ++r) {
      const int qrow = q0 + r + 8 * hg;
      const int key0 = ks + col;
      const int key1 = ks + 16 + col;
      float v0 = (key0 <= qrow) ? s0[r] * scale : -INFINITY;
      float v1 = (key1 <= qrow) ? s1[r] * scale : -INFINITY;
      float mx = fmaxf(v0, v1);
#pragma unroll
      for (int off = 1; off < 16; off <<= 1) mx = fmaxf(mx, __shfl_xor(mx, off, 16));
      const float mnew = fmaxf(mrow[r], mx);
      const float alpha = __expf(mrow[r] - mnew);
      const float p0 = __expf(v0 - mnew);
      const float p1 = __expf(v1 - mnew);
      float rs = p0 + p1;
#pragma unroll
      for (int off = 1; off < 16; off <<= 1) rs += __shfl_xor(rs, off, 16);
      lrow[r] = lrow[r] * alpha + rs;
      mrow[r] = mnew;
      alphar[r] = alpha;
      const int row = r + 8 * hg;
      pshm[wave][row][col] = (bf16)p0;
      pshm[wave][row][col + 16] = (bf16)p1;
    }
    // rescale running output
#pragma unroll
    for (int j = 0; j < 8; ++j)
#pragma unroll
      for (int r = 0; r < 8; ++r) o[j][r] *= alphar[r];

    // P back as an A-fragment (same-wave LDS ops are in-order on CDNA5)
    v16bf pf = load_frag(&pshm[wave][0][0], 32);

    // ---- PV: accumulate 16x128 ----
#pragma unroll
    for (int j = 0; j < 8; ++j) {
      v16bf vf = load_frag(Vh + (size_t)(j * 16) * kS + ks, kS);
      o[j] = WMMA_BF16(pf, vf, o[j]);
    }
  }

  // epilogue: divide by row sums, store (b,s,h,d) as token-major f32
#pragma unroll
  for (int j = 0; j < 8; ++j) {
#pragma unroll
    for (int r = 0; r < 8; ++r) {
      const int qrow = q0 + r + 8 * hg;
      const int d = j * 16 + col;
      O[((size_t)(b * kS + qrow)) * kH + (size_t)h * kHD + d] = o[j][r] / lrow[r];
    }
  }
}

// ---------------------------------------------------------------------------
// Elementwise / small kernels
// ---------------------------------------------------------------------------
__global__ void cast_f32_bf16_kernel(const float* __restrict__ in, bf16* __restrict__ out, long n) {
  long i = (long)blockIdx.x * blockDim.x + threadIdx.x;
  if (i < n) out[i] = (bf16)in[i];
}

__global__ void __launch_bounds__(256)
rmsnorm_kernel(const float* __restrict__ x, const float* __restrict__ w, bf16* __restrict__ out) {
  const int row = blockIdx.x;
  const float* xr = x + (size_t)row * kH;
  float ss = 0.f;
  for (int i = threadIdx.x; i < kH; i += 256) { float v = xr[i]; ss += v * v; }
#pragma unroll
  for (int off = 16; off; off >>= 1) ss += __shfl_xor(ss, off, 32);
  __shared__ float red[8];
  __shared__ float stot;
  if ((threadIdx.x & 31) == 0) red[threadIdx.x >> 5] = ss;
  __syncthreads();
  if (threadIdx.x == 0) {
    float t = 0.f;
    for (int i = 0; i < 8; ++i) t += red[i];
    stot = rsqrtf(t / (float)kH + 1e-6f);
  }
  __syncthreads();
  const float rs = stot;
  for (int i = threadIdx.x; i < kH; i += 256)
    out[(size_t)row * kH + i] = (bf16)(xr[i] * rs * w[i]);
}

__global__ void rope_kernel(const float* __restrict__ qin, const float* __restrict__ kin,
                            const float* __restrict__ cosb, const float* __restrict__ sinb,
                            bf16* __restrict__ qo, bf16* __restrict__ ko) {
  long idx = (long)blockIdx.x * blockDim.x + threadIdx.x;    // (t, h, d<64)
  const long total = (long)kT * kNH * (kHD / 2);
  if (idx >= total) return;
  const int d = (int)(idx & 63);
  const int h = (int)((idx >> 6) & (kNH - 1));
  const long t = idx >> 9;
  const int s = (int)(t & (kS - 1));
  const size_t base = (size_t)t * kH + (size_t)h * kHD;
  const float c1 = cosb[(size_t)s * kHD + d],      sn1 = sinb[(size_t)s * kHD + d];
  const float c2 = cosb[(size_t)s * kHD + d + 64], sn2 = sinb[(size_t)s * kHD + d + 64];
  const float q1 = qin[base + d], q2 = qin[base + d + 64];
  const float k1 = kin[base + d], k2 = kin[base + d + 64];
  qo[base + d]      = (bf16)(q1 * c1 - q2 * sn1);
  qo[base + d + 64] = (bf16)(q2 * c2 + q1 * sn2);
  ko[base + d]      = (bf16)(k1 * c1 - k2 * sn1);
  ko[base + d + 64] = (bf16)(k2 * c2 + k1 * sn2);
}

__global__ void vtranspose_kernel(const float* __restrict__ v, bf16* __restrict__ vt) {
  long idx = (long)blockIdx.x * blockDim.x + threadIdx.x;    // (b,h,d,s)
  const long total = (long)kB * kNH * kHD * kS;
  if (idx >= total) return;
  const int s = (int)(idx & (kS - 1));
  long r = idx >> 10;
  const int d = (int)(r & (kHD - 1));
  r >>= 7;
  const int h = (int)(r & (kNH - 1));
  const int b = (int)(r >> 3);
  vt[idx] = (bf16)v[((size_t)(b * kS + s)) * kH + (size_t)h * kHD + d];
}

__global__ void __launch_bounds__(256)
gate_logits_kernel(const bf16* __restrict__ xn, const bf16* __restrict__ gw,
                   float* __restrict__ logits) {
  const int t = blockIdx.x;
  const int e = threadIdx.x >> 5;     // 8 waves, one expert each
  const int lane = threadIdx.x & 31;
  const bf16* xr = xn + (size_t)t * kH;
  const bf16* wr = gw + (size_t)e * kH;
  float s = 0.f;
  for (int i = lane; i < kH; i += 32) s += (float)xr[i] * (float)wr[i];
#pragma unroll
  for (int off = 16; off; off >>= 1) s += __shfl_xor(s, off, 32);
  if (lane == 0) logits[(size_t)t * kE + e] = s;
}

__global__ void zero_i32_kernel(int* p, int n) {
  int i = blockIdx.x * blockDim.x + threadIdx.x;
  if (i < n) p[i] = 0;
}

__global__ void zero_f32_kernel(float* p, long n) {
  long i = (long)blockIdx.x * blockDim.x + threadIdx.x;
  if (i < n) p[i] = 0.f;
}

__global__ void route_topk_kernel(const float* __restrict__ logits, int* __restrict__ cnt,
                                  int* __restrict__ list, float* __restrict__ wts) {
  const int t = blockIdx.x * blockDim.x + threadIdx.x;
  if (t >= kT) return;
  float l[kE];
  float mx = -INFINITY;
#pragma unroll
  for (int e = 0; e < kE; ++e) { l[e] = logits[(size_t)t * kE + e]; mx = fmaxf(mx, l[e]); }
  float se = 0.f;
#pragma unroll
  for (int e = 0; e < kE; ++e) { l[e] = __expf(l[e] - mx); se += l[e]; }
  int i0 = 0;
#pragma unroll
  for (int e = 1; e < kE; ++e) if (l[e] > l[i0]) i0 = e;
  int i1 = (i0 == 0) ? 1 : 0;
#pragma unroll
  for (int e = 0; e < kE; ++e) if (e != i0 && l[e] > l[i1]) i1 = e;
  float p0 = l[i0] / se, p1 = l[i1] / se;
  const float s2 = p0 + p1 + 1e-20f;
  p0 /= s2; p1 /= s2;
  int pos0 = atomicAdd(&cnt[i0], 1);
  list[(size_t)i0 * kT + pos0] = t;
  wts[(size_t)i0 * kT + pos0] = p0;
  int pos1 = atomicAdd(&cnt[i1], 1);
  list[(size_t)i1 * kT + pos1] = t;
  wts[(size_t)i1 * kT + pos1] = p1;
}

__global__ void __launch_bounds__(256)
gather_rows_kernel(const bf16* __restrict__ xn, const int* __restrict__ list,
                   const int* __restrict__ cnt, bf16* __restrict__ Ag, int e) {
  const int i = blockIdx.x;
  if (i >= cnt[e]) return;
  const int t = list[(size_t)e * kT + i];
  for (int d = threadIdx.x; d < kH; d += 256)
    Ag[(size_t)i * kH + d] = xn[(size_t)t * kH + d];
}

__global__ void silu_mul_kernel(const float* __restrict__ g, const float* __restrict__ u,
                                bf16* __restrict__ h, long n) {
  long i = (long)blockIdx.x * blockDim.x + threadIdx.x;
  if (i >= n) return;
  const float gv = g[i];
  const float s = gv / (1.f + __expf(-gv));
  h[i] = (bf16)(s * u[i]);
}

__global__ void __launch_bounds__(256)
scatter_add_kernel(const float* __restrict__ y, const int* __restrict__ list,
                   const float* __restrict__ wts, const int* __restrict__ cnt,
                   float* __restrict__ routed, int e) {
  const int i = blockIdx.x;
  if (i >= cnt[e]) return;
  const int t = list[(size_t)e * kT + i];
  const float w = wts[(size_t)e * kT + i];
  for (int d = threadIdx.x; d < kH; d += 256)
    routed[(size_t)t * kH + d] += w * y[(size_t)i * kH + d];
}

__global__ void add3_kernel(const float* __restrict__ a, const float* __restrict__ b,
                            const float* __restrict__ c, float* __restrict__ out, long n) {
  long i = (long)blockIdx.x * blockDim.x + threadIdx.x;
  if (i < n) out[i] = a[i] + b[i] + c[i];
}

// ---------------------------------------------------------------------------
// Host-side orchestration
// ---------------------------------------------------------------------------
static inline dim3 gemm_grid(int M, int N) { return dim3((N + 127) / 128, (M + 127) / 128); }

extern "C" void kernel_launch(void* const* d_in, const int* in_sizes, int n_in,
                              void* d_out, int out_size, void* d_ws, size_t ws_size,
                              hipStream_t stream) {
  (void)in_sizes; (void)n_in; (void)out_size; (void)ws_size;

  const float* hidden = (const float*)d_in[0];
  const float* ln1    = (const float*)d_in[1];
  const float* ln2    = (const float*)d_in[2];
  const float* qw     = (const float*)d_in[3];
  const float* kw     = (const float*)d_in[4];
  const float* vw     = (const float*)d_in[5];
  const float* ow     = (const float*)d_in[6];
  const float* cosb   = (const float*)d_in[7];
  const float* sinb   = (const float*)d_in[8];
  const float* gatew  = (const float*)d_in[9];
  const float* egw    = (const float*)d_in[10];
  const float* euw    = (const float*)d_in[11];
  const float* edw    = (const float*)d_in[12];
  const float* sgw    = (const float*)d_in[13];
  const float* suw    = (const float*)d_in[14];
  const float* sdw    = (const float*)d_in[15];
  float* out = (float*)d_out;

  // Workspace carve-out
  char* p = (char*)d_ws;
  auto alloc = [&](size_t bytes) -> void* {
    void* r = (void*)p;
    p += (bytes + 255) & ~(size_t)255;
    return r;
  };
  const size_t TH = (size_t)kT * kH;

  bf16* wq_b  = (bf16*)alloc((size_t)kH * kH * 2);
  bf16* wk_b  = (bf16*)alloc((size_t)kH * kH * 2);
  bf16* wv_b  = (bf16*)alloc((size_t)kH * kH * 2);
  bf16* wo_b  = (bf16*)alloc((size_t)kH * kH * 2);
  bf16* weg_b = (bf16*)alloc((size_t)kE * kMI * kH * 2);
  bf16* weu_b = (bf16*)alloc((size_t)kE * kMI * kH * 2);
  bf16* wed_b = (bf16*)alloc((size_t)kE * kH * kMI * 2);
  bf16* wsg_b = (bf16*)alloc((size_t)kSI * kH * 2);
  bf16* wsu_b = (bf16*)alloc((size_t)kSI * kH * 2);
  bf16* wsd_b = (bf16*)alloc((size_t)kH * kSI * 2);
  bf16* wg_b  = (bf16*)alloc((size_t)kE * kH * 2);

  bf16*  xn1  = (bf16*) alloc(TH * 2);
  float* qf   = (float*)alloc(TH * 4);
  float* kf   = (float*)alloc(TH * 4);
  float* vf   = (float*)alloc(TH * 4);
  bf16*  qb   = (bf16*) alloc(TH * 2);
  bf16*  kb   = (bf16*) alloc(TH * 2);
  bf16*  vt   = (bf16*) alloc(TH * 2);
  float* of   = (float*)alloc(TH * 4);
  bf16*  ob   = (bf16*) alloc(TH * 2);
  float* x1   = (float*)alloc(TH * 4);
  bf16*  xn2  = (bf16*) alloc(TH * 2);

  float* logits = (float*)alloc((size_t)kT * kE * 4);
  int*   cnt    = (int*)  alloc((size_t)kE * 4);
  int*   list   = (int*)  alloc((size_t)kE * kT * 4);
  float* wts    = (float*)alloc((size_t)kE * kT * 4);
  bf16*  Ag     = (bf16*) alloc(TH * 2);
  float* gf     = (float*)alloc((size_t)kT * kMI * 4);
  float* uf     = (float*)alloc((size_t)kT * kMI * 4);
  bf16*  hb     = (bf16*) alloc((size_t)kT * kMI * 2);
  float* yf     = (float*)alloc(TH * 4);
  float* routed = (float*)alloc(TH * 4);
  float* sgf    = (float*)alloc((size_t)kT * kSI * 4);
  float* suf    = (float*)alloc((size_t)kT * kSI * 4);
  bf16*  hsb    = (bf16*) alloc((size_t)kT * kSI * 2);
  float* shrd   = (float*)alloc(TH * 4);

  auto cast = [&](const float* src, bf16* dst, long n) {
    cast_f32_bf16_kernel<<<(int)((n + 255) / 256), 256, 0, stream>>>(src, dst, n);
  };

  // --- weight casts (per-call; deterministic) ---
  cast(qw, wq_b, (long)kH * kH);
  cast(kw, wk_b, (long)kH * kH);
  cast(vw, wv_b, (long)kH * kH);
  cast(ow, wo_b, (long)kH * kH);
  cast(egw, weg_b, (long)kE * kMI * kH);
  cast(euw, weu_b, (long)kE * kMI * kH);
  cast(edw, wed_b, (long)kE * kH * kMI);
  cast(sgw, wsg_b, (long)kSI * kH);
  cast(suw, wsu_b, (long)kSI * kH);
  cast(sdw, wsd_b, (long)kH * kSI);
  cast(gatew, wg_b, (long)kE * kH);

  // --- attention ---
  rmsnorm_kernel<<<kT, 256, 0, stream>>>(hidden, ln1, xn1);
  gemm_bf16_kernel<<<gemm_grid(kT, kH), 256, 0, stream>>>(xn1, wq_b, qf, nullptr, kT, kH, kH, nullptr);
  gemm_bf16_kernel<<<gemm_grid(kT, kH), 256, 0, stream>>>(xn1, wk_b, kf, nullptr, kT, kH, kH, nullptr);
  gemm_bf16_kernel<<<gemm_grid(kT, kH), 256, 0, stream>>>(xn1, wv_b, vf, nullptr, kT, kH, kH, nullptr);

  {
    const long n = (long)kT * kNH * (kHD / 2);
    rope_kernel<<<(int)((n + 255) / 256), 256, 0, stream>>>(qf, kf, cosb, sinb, qb, kb);
  }
  {
    const long n = (long)kB * kNH * kHD * kS;
    vtranspose_kernel<<<(int)((n + 255) / 256), 256, 0, stream>>>(vf, vt);
  }
  flash_attn_kernel<<<dim3(kS / 128, kNH, kB), 256, 0, stream>>>(qb, kb, vt, of);
  cast(of, ob, (long)TH);
  gemm_bf16_kernel<<<gemm_grid(kT, kH), 256, 0, stream>>>(ob, wo_b, x1, hidden, kT, kH, kH, nullptr);

  // --- MoE ---
  rmsnorm_kernel<<<kT, 256, 0, stream>>>(x1, ln2, xn2);
  gate_logits_kernel<<<kT, 256, 0, stream>>>(xn2, wg_b, logits);
  zero_i32_kernel<<<1, kE, 0, stream>>>(cnt, kE);
  zero_f32_kernel<<<(int)((TH + 255) / 256), 256, 0, stream>>>(routed, (long)TH);
  route_topk_kernel<<<(kT + 255) / 256, 256, 0, stream>>>(logits, cnt, list, wts);

  for (int e = 0; e < kE; ++e) {
    const int* ce = cnt + e;
    gather_rows_kernel<<<kT, 256, 0, stream>>>(xn2, list, cnt, Ag, e);
    gemm_bf16_kernel<<<gemm_grid(kT, kMI), 256, 0, stream>>>(
        Ag, weg_b + (size_t)e * kMI * kH, gf, nullptr, kT, kMI, kH, ce);
    gemm_bf16_kernel<<<gemm_grid(kT, kMI), 256, 0, stream>>>(
        Ag, weu_b + (size_t)e * kMI * kH, uf, nullptr, kT, kMI, kH, ce);
    silu_mul_kernel<<<(int)(((long)kT * kMI + 255) / 256), 256, 0, stream>>>(
        gf, uf, hb, (long)kT * kMI);
    gemm_bf16_kernel<<<gemm_grid(kT, kH), 256, 0, stream>>>(
        hb, wed_b + (size_t)e * kH * kMI, yf, nullptr, kT, kH, kMI, ce);
    scatter_add_kernel<<<kT, 256, 0, stream>>>(yf, list, wts, cnt, routed, e);
  }

  // shared expert MLP
  gemm_bf16_kernel<<<gemm_grid(kT, kSI), 256, 0, stream>>>(xn2, wsg_b, sgf, nullptr, kT, kSI, kH, nullptr);
  gemm_bf16_kernel<<<gemm_grid(kT, kSI), 256, 0, stream>>>(xn2, wsu_b, suf, nullptr, kT, kSI, kH, nullptr);
  silu_mul_kernel<<<(int)(((long)kT * kSI + 255) / 256), 256, 0, stream>>>(
      sgf, suf, hsb, (long)kT * kSI);
  gemm_bf16_kernel<<<gemm_grid(kT, kH), 256, 0, stream>>>(hsb, wsd_b, shrd, nullptr, kT, kH, kSI, nullptr);

  // final residual combine
  add3_kernel<<<(int)((TH + 255) / 256), 256, 0, stream>>>(x1, routed, shrd, out, (long)TH);
}
